// Codebook_10136122819267
// MI455X (gfx1250) — matrix operand order repaired and checked by reading, not compiled, not run on hardware
//
#include <hip/hip_runtime.h>
#include <hip/hip_bf16.h>
#include <math.h>

// ---------------------------------------------------------------------------
// VQ codebook lookup for MI455X (gfx1250, wave32, WMMA).
// z: (4,256,8,32,32) f32, embeddings: (2048,256) f32.
// Scores: ||e_j||^2 - 2*x.e_j via bf16 WMMA (f32 accumulate).
// Block = 256 threads (8 waves), M=128 rows; B tile shared via LDS double
// buffer (8x L2 traffic reduction vs per-wave streaming), async LDS fill.
// ---------------------------------------------------------------------------

typedef __bf16        v16bf __attribute__((ext_vector_type(16)));
typedef float         v8f   __attribute__((ext_vector_type(8)));
typedef unsigned int  u32x4 __attribute__((ext_vector_type(4)));
typedef unsigned int  u32x8 __attribute__((ext_vector_type(8)));

#define NCODES   2048
#define DCH      256
#define NVEC     32768          // 4*8*32*32
#define SPB      8192           // T*H*W per batch
#define ZPB      2097152        // 256*8192 per batch
#define EMB_OUT_ELEMS 8388608   // 4*256*8192
#define MTILE    128            // rows per block
#define NJT      (NCODES / 16)  // 128 codebook N-tiles

#if __has_builtin(__builtin_amdgcn_global_load_async_to_lds_b128)
#define HAVE_ASYNC_LDS 1
#else
#define HAVE_ASYNC_LDS 0
#endif

// Builtin signature (from hipcc diagnostic): first param is a pointer to a
// GCC-vector int4 in the global AS, second in the LDS AS, then imm offset+cpol.
typedef int gv4i __attribute__((vector_size(16)));
using as1_v4i = __attribute__((address_space(1))) gv4i;
using as3_v4i = __attribute__((address_space(3))) gv4i;

__device__ __forceinline__ void wait_async0() {
#if __has_builtin(__builtin_amdgcn_s_wait_asynccnt)
    __builtin_amdgcn_s_wait_asynccnt(0);
#elif HAVE_ASYNC_LDS
    asm volatile("s_wait_asynccnt 0x0" ::: "memory");
#endif
}

// Two 16B chunks -> one 16-element bf16 fragment register block.
__device__ __forceinline__ v16bf load_frag(const void* p0, const void* p1) {
    u32x4 a = *(const u32x4*)p0;
    u32x4 b = *(const u32x4*)p1;
    u32x8 c = __builtin_shufflevector(a, b, 0, 1, 2, 3, 4, 5, 6, 7);
    return __builtin_bit_cast(v16bf, c);
}

// ---------------------------------------------------------------------------
// Kernel A: codebook prep (bf16 copy, f32 norms, zero accumulators).
// ---------------------------------------------------------------------------
__global__ __launch_bounds__(256) void vq_prep_kernel(
    const float* __restrict__ emb, __bf16* __restrict__ embbf,
    float* __restrict__ enorm, float* __restrict__ counts,
    float* __restrict__ lossacc) {
    const int j = blockIdx.x;
    const int c = threadIdx.x;
    const float v = emb[j * DCH + c];
    embbf[j * DCH + c] = (__bf16)v;

    __shared__ float sred[256];
    sred[c] = v * v;
    __syncthreads();
    for (int o = 128; o > 0; o >>= 1) {
        if (c < o) sred[c] += sred[c + o];
        __syncthreads();
    }
    if (c == 0) {
        enorm[j]  = sred[0];
        counts[j] = 0.0f;
        if (j == 0) lossacc[0] = 0.0f;
    }
}

// ---------------------------------------------------------------------------
// Kernel B: main tile kernel. 256 blocks x 256 threads (8 waves).
// ---------------------------------------------------------------------------
__global__ __launch_bounds__(256) void vq_main_kernel(
    const float* __restrict__ z, const float* __restrict__ emb,
    const __bf16* __restrict__ embbf, const float* __restrict__ enorm,
    float* __restrict__ counts, float* __restrict__ lossacc,
    float* __restrict__ outEmb, float* __restrict__ outIdx) {
    __shared__ __bf16 Ascratch[16][DCH + 8];      //  8448 B (startup only)
    __shared__ __bf16 Btile[2][16][DCH + 8];      // 16896 B double buffer
    __shared__ float  enormS[NCODES];             //  8192 B
    __shared__ int    fidxS[MTILE];               //   512 B
    __shared__ float  lred[256];                  //  1024 B

    const int tid  = threadIdx.x;
    const int lane = tid & 31;
    const int wv   = tid >> 5;                    // wave id 0..7 (owns 16 rows)
    const int m    = lane & 15;
    const int half = lane >> 4;
    const int n0   = blockIdx.x * MTILE;          // first of 128 rows
    const int b    = n0 >> 13;
    const int sp0  = n0 & (SPB - 1);
    const size_t zbase = (size_t)b * ZPB + (size_t)sp0;   // + c*8192 + row

    // ---- Stage ||e||^2 into LDS ------------------------------------------
    for (int j0 = tid; j0 < NCODES; j0 += 256) enormS[j0] = enorm[j0];

    // ---- Build the 8 per-wave A fragment sets through one LDS scratch ----
    // 16-bit A 16x32 layout: lanes 0-15 hold K {0..7,16..23},
    //                        lanes 16-31 hold K {8..15,24..31}.
    v16bf af[8];
    for (int st = 0; st < 8; ++st) {
        __syncthreads();
        {   // all 256 threads stage rows [st*16, st*16+16) x 256 channels
            const int row = tid & 15;
            const int grp = tid >> 4;             // 0..15
            #pragma unroll
            for (int i = 0; i < 16; ++i) {
                const int c = grp + (i << 4);     // coalesced 64B per 16 lanes
                Ascratch[row][c] =
                    (__bf16)z[zbase + (size_t)c * SPB + (st << 4) + row];
            }
        }
        __syncthreads();
        if (wv == st) {
            #pragma unroll
            for (int kk = 0; kk < 8; ++kk) {
                const int k0 = kk << 5;
                af[kk] = load_frag(&Ascratch[m][k0 + half * 8],
                                   &Ascratch[m][k0 + 16 + half * 8]);
            }
        }
    }

    // ---- B tile staging helper: 8KB tile, 32B per thread -----------------
    const int trow = tid >> 4;                    // code row within tile 0..15
    const int tcol = (tid & 15) << 4;             // bf16 column 0..240
    auto stage_b = [&](int jt, int buf) {
        const __bf16* g = embbf + (size_t)((jt << 4) + trow) * DCH + tcol;
        __bf16* l = &Btile[buf][trow][tcol];
#if HAVE_ASYNC_LDS
        __builtin_amdgcn_global_load_async_to_lds_b128(
            (as1_v4i*)g, (as3_v4i*)l, 0, 0);
        __builtin_amdgcn_global_load_async_to_lds_b128(
            (as1_v4i*)(g + 8), (as3_v4i*)(l + 8), 0, 0);
#else
        u32x4 d0 = *(const u32x4*)g;
        u32x4 d1 = *(const u32x4*)(g + 8);
        *(u32x4*)&Btile[buf][trow][tcol]     = d0;
        *(u32x4*)&Btile[buf][trow][tcol + 8] = d1;
#endif
    };

    float mval[8];
    int   midx[8];
    #pragma unroll
    for (int r = 0; r < 8; ++r) { mval[r] = 3.4e38f; midx[r] = 0; }

    __syncthreads();
    stage_b(0, 0);

    // ---- Main sweep: all 8 waves consume the shared B tile ---------------
    // 16-bit B 32x16 layout: lane n = column n; lanes 0-15 K {0..7,16..23},
    // lanes 16-31 K {8..15,24..31} -> two 16B chunks per lane per k-step.
    for (int jt = 0; jt < NJT; ++jt) {
        const int buf = jt & 1;
        wait_async0();                  // own async fills for tile jt done
        __syncthreads();                // everyone's fills done; buf^1 free
        if (jt + 1 < NJT) stage_b(jt + 1, buf ^ 1);

        const int j = (jt << 4) + m;    // code id / column
        const __bf16* bbase = &Btile[buf][m][half * 8];
        v8f acc = {};
        #pragma unroll
        for (int kk = 0; kk < 8; ++kk) {
            v16bf bf = load_frag(bbase + (kk << 5), bbase + (kk << 5) + 16);
            acc = __builtin_amdgcn_wmma_f32_16x16x32_bf16(
                false, af[kk], false, bf, (short)0, acc, false, false);
        }
        const float en = enormS[j];
        // C/D layout: VGPR r <-> row (r + half*8), column j.
        #pragma unroll
        for (int r = 0; r < 8; ++r) {
            const float d = en - 2.0f * acc[r];
            if (d < mval[r]) { mval[r] = d; midx[r] = j; }
        }
    }

    // ---- Argmin reduce across the 16 columns in each 16-lane half --------
    #pragma unroll
    for (int r = 0; r < 8; ++r) {
        #pragma unroll
        for (int mk = 8; mk >= 1; mk >>= 1) {
            const float ov = __shfl_xor(mval[r], mk, 16);
            const int   oi = __shfl_xor(midx[r], mk, 16);
            if (ov < mval[r] || (ov == mval[r] && oi < midx[r])) {
                mval[r] = ov; midx[r] = oi;
            }
        }
    }
    if (m == 0) {                 // lane 0 -> rows +0..7, lane 16 -> rows +8..15
        const int rb = (wv << 4) + (half << 3);
        #pragma unroll
        for (int r = 0; r < 8; ++r) fidxS[rb + r] = midx[r];
    }
    __syncthreads();

    if (tid < MTILE) {
        const int bi = fidxS[tid];
        outIdx[n0 + tid] = (float)bi;
        atomicAdd(&counts[bi], 1.0f);
    }

    // ---- Gather winning codes, straight-through output, MSE partial ------
    float ls = 0.0f;
    {
        const int row = tid & 15;
        const int grp = tid >> 4;             // 0..15 channel groups of 16
        for (int st = 0; st < 8; ++st) {
            const float* evec = emb + (size_t)fidxS[(st << 4) + row] * DCH;
            #pragma unroll
            for (int i = 0; i < 16; ++i) {
                const int c = (grp << 4) + i;
                const float ev = evec[c];
                const size_t a = zbase + (size_t)c * SPB + (st << 4) + row;
                outEmb[a] = ev;               // coalesced 64B per 16 lanes
                const float df = z[a] - ev;
                ls += df * df;
            }
        }
    }
    lred[tid] = ls;
    __syncthreads();
    for (int o = 128; o > 0; o >>= 1) {
        if (tid < o) lred[tid] += lred[tid + o];
        __syncthreads();
    }
    if (tid == 0) atomicAdd(lossacc, lred[0]);
}

// ---------------------------------------------------------------------------
// Kernel C: scalars (commitment loss, perplexity).
// ---------------------------------------------------------------------------
__global__ __launch_bounds__(256) void vq_final_kernel(
    const float* __restrict__ counts, const float* __restrict__ lossacc,
    float* __restrict__ out) {
    __shared__ float sred[256];
    float s = 0.0f;
    for (int j = threadIdx.x; j < NCODES; j += 256) {
        const float p = counts[j] * (1.0f / (float)NVEC);
        s += p * logf(p + 1e-10f);
    }
    sred[threadIdx.x] = s;
    __syncthreads();
    for (int o = 128; o > 0; o >>= 1) {
        if (threadIdx.x < o) sred[threadIdx.x] += sred[threadIdx.x + o];
        __syncthreads();
    }
    if (threadIdx.x == 0) {
        out[EMB_OUT_ELEMS + NVEC]     = 0.25f * lossacc[0] / (float)EMB_OUT_ELEMS;
        out[EMB_OUT_ELEMS + NVEC + 1] = expf(-sred[0]);
    }
}

// ---------------------------------------------------------------------------
// Launch wrapper.
// d_in[0] = z (f32, 8388608), d_in[1] = embeddings (f32, 524288)
// d_out   = [emb_st (8388608) | indices (32768) | loss (1) | perplexity (1)]
// d_ws    = [bf16 codebook 1MB | enorm 8KB | counts 8KB | loss acc]
// ---------------------------------------------------------------------------
extern "C" void kernel_launch(void* const* d_in, const int* in_sizes, int n_in,
                              void* d_out, int out_size, void* d_ws, size_t ws_size,
                              hipStream_t stream) {
    (void)in_sizes; (void)n_in; (void)out_size; (void)ws_size;
    const float* z   = (const float*)d_in[0];
    const float* emb = (const float*)d_in[1];

    char*   ws      = (char*)d_ws;
    __bf16* embbf   = (__bf16*)ws;                                  // 1,048,576 B
    float*  enorm   = (float*)(ws + 1048576);                       // 8,192 B
    float*  counts  = (float*)(ws + 1048576 + 8192);                // 8,192 B
    float*  lossacc = (float*)(ws + 1048576 + 16384);               // 4 B

    float* out    = (float*)d_out;
    float* outEmb = out;
    float* outIdx = out + EMB_OUT_ELEMS;

    vq_prep_kernel<<<NCODES, 256, 0, stream>>>(emb, embbf, enorm, counts, lossacc);
    vq_main_kernel<<<NVEC / MTILE, 256, 0, stream>>>(z, emb, embbf, enorm, counts,
                                                     lossacc, outEmb, outIdx);
    vq_final_kernel<<<1, 256, 0, stream>>>(counts, lossacc, out);
}
